// Model_2757369004057
// MI455X (gfx1250) — matrix operand order repaired and checked by reading, not compiled, not run on hardware
//
#include <hip/hip_runtime.h>

// ---------------------------------------------------------------------------
// CDNA5 / gfx1250 persistent RNN:
//  - v_wmma_f32_16x16x32_bf16 for the recurrent GEMM (W_eff L2-resident, 8MB bf16)
//  - W_in / b_rec folded into an extra K-tile of the same WMMA stream
//  - A fragments LDS-staged per WG per step; W streamed from L2
//  - K loop chunked by 4: one wait per 4 WMMAs instead of per WMMA
//  - persistent kernel + device-wide atomic barrier (1000 dependent steps)
// ---------------------------------------------------------------------------

typedef __bf16 bf16;
typedef __attribute__((ext_vector_type(16))) __bf16 v16bf;
typedef __attribute__((ext_vector_type(8)))  __bf16 v8bf;
typedef __attribute__((ext_vector_type(8)))  float  v8f;

#define HID    512
#define NN     2048
#define BB     32
#define TT     1000
#define INPD   16
#define UNITS  100
#define ALPHA_F      0.1f
#define ONE_M_ALPHA  0.9f
#define INP_NOISE_F  0.01f
#define ACT_NOISE_F  0.01f

#define NWG        32           // persistent grid (all resident)
#define KT_N       64           // 32-wide K tiles covering N=2048
#define KT_TOT     65           // + drive tile (u, bias)
#define FRAG_ELEMS 512          // 32 lanes * 16 elems per 32x16 bf16 B / 16x32 A frag
#define OUT_ELEMS  (BB*TT*400)

// d_ws layout (bytes)
#define OFF_CNT   0
#define OFF_HB    1024
#define HB_ELEMS  (2*KT_N*FRAG_ELEMS)          // one h ping buffer: 65536 bf16
#define OFF_UB    (OFF_HB + 2*HB_ELEMS*2)      // 263168
#define UB_ELEMS  (2*FRAG_ELEMS)               // u/bias frag per mt
#define OFF_RPK   (OFF_UB + UB_ELEMS*2 + 1024) // 266240
#define RPK_ELEMS (4*7*16*FRAG_ELEMS)          // 229376 bf16
#define CLEAN_BYTES OFF_RPK

static __device__ __forceinline__ v8f wmma_bf16(v16bf a, v16bf b, v8f c) {
  return __builtin_amdgcn_wmma_f32_16x16x32_bf16(false, a, false, b,
                                                 (short)0, c, false, false);
}

// ---- pack W_eff (+W_in, +b_rec) into per-fragment bf16 B layout -------------
__global__ void prep_w(const float* __restrict__ W_raw, const float* __restrict__ W_in,
                       const float* __restrict__ b_rec, bf16* __restrict__ wpk) {
  int idx = blockIdx.x * blockDim.x + threadIdx.x;     // one thread per (nt,kt,lane)
  if (idx >= 128 * KT_TOT * 32) return;
  int lane = idx & 31;
  int rest = idx >> 5;
  int kt = rest % KT_TOT, nt = rest / KT_TOT;
  int nl = lane & 15, kh = lane >> 4;
  int n = nt * 16 + nl;
  v16bf v;
  if (kt < KT_N) {
    int k0 = kt * 32 + kh * 8;
#pragma unroll
    for (int e = 0; e < 16; ++e) {
      int k = k0 + (e & 7) + ((e >> 3) << 4);
      float w = fmaxf(W_raw[(size_t)n * NN + k], 0.f);
      if ((k & 511) >= 384) w = -w;                    // inhibitory columns
      v[e] = (bf16)w;
    }
  } else {                                             // drive tile: W_in row, b_rec, 0
#pragma unroll
    for (int e = 0; e < 16; ++e) {
      int ke = kh * 8 + (e & 7) + ((e >> 3) << 4);
      float w = (ke < INPD) ? W_in[n * INPD + ke] : (ke == INPD ? b_rec[n] : 0.f);
      v[e] = (bf16)w;
    }
  }
  *(v16bf*)(wpk + (size_t)idx * 16) = v;
}

// ---- pack the 4 readout weight matrices (bla permutation folded in) ---------
__global__ void prep_r(const float* __restrict__ Wp, const float* __restrict__ Wa,
                       const float* __restrict__ Wo, const float* __restrict__ Wb,
                       bf16* __restrict__ rpk) {
  int idx = blockIdx.x * blockDim.x + threadIdx.x;     // one per (rr,ntile,kt,lane)
  if (idx >= 4 * 7 * 16 * 32) return;
  int lane = idx & 31;
  int rest = idx >> 5;
  int kt = rest & 15; rest >>= 4;
  int ntile = rest % 7, rr = rest / 7;
  int nl = lane & 15, kh = lane >> 4;
  int n = ntile * 16 + nl;
  v16bf v;
#pragma unroll
  for (int e = 0; e < 16; ++e) {
    int k = kt * 32 + kh * 8 + (e & 7) + ((e >> 3) << 4); // 0..511 within region slice
    float w = 0.f;
    if (n < UNITS) {
      if (rr == 0)      w = Wp[n * HID + k];
      else if (rr == 1) w = Wa[n * HID + k];
      else if (rr == 2) w = Wo[n * HID + k];
      else { int wk = (k < 128) ? (384 + k) : (k - 128); w = Wb[n * HID + wk]; }
    }
    v[e] = (bf16)w;
  }
  *(v16bf*)(rpk + (size_t)idx * 16) = v;
}

// ---- initial h = relu(xn) into ping buffer 0; constant-1 bias lane ----------
__global__ void init_h(const float* __restrict__ xn, bf16* __restrict__ hb,
                       bf16* __restrict__ ub) {
  int gid = blockIdx.x * blockDim.x + threadIdx.x;
  if (gid < BB * NN) {
    int b = gid >> 11, n = gid & (NN - 1);
    float h0 = fmaxf(xn[gid], 0.f);
    int mt = b >> 4, kt = n >> 5, kl = n & 31;
    int half = (kl >> 3) & 1;
    int lanew = (b & 15) + (half << 4);
    int elem = (kl & 7) + ((kl >> 4) << 3);
    hb[((size_t)(mt * KT_N + kt) * 32 + lanew) * 16 + elem] = (bf16)h0;
  }
  if (gid < BB) {                                       // A[m][K=16] = 1 (bias column)
    int b = gid;
    ub[(((b >> 4) * 32) + (b & 15)) * 16 + 8] = (bf16)1.0f;
  }
}

// ---- persistent recurrence: 32 WGs x 8 waves, one 16x16 tile per wave -------
__global__ void __launch_bounds__(256)
rnn_recurrent(const float* __restrict__ xn, const float* __restrict__ inp,
              const float* __restrict__ eps_inp, const float* __restrict__ eps_act,
              const bf16* __restrict__ wpk, float* __restrict__ hn,
              bf16* __restrict__ hb, bf16* __restrict__ ub,
              unsigned* __restrict__ cnt) {
  extern __shared__ bf16 As[];                    // 64 KB: this WG's A slice (one mt)
  const int tid = threadIdx.x, lane = tid & 31, wave = tid >> 5, wg = blockIdx.x;
  const int mt = wg >> 4;                         // batch tile (0..1)
  const int nt = ((wg & 15) << 3) + wave;         // column tile (0..127)
  const int nl = lane & 15, mh = lane >> 4;
  const int n = nt * 16 + nl;

  float x[8];
  int brow[8];
#pragma unroll
  for (int r = 0; r < 8; ++r) {
    brow[r] = mt * 16 + r + (mh << 3);            // C layout: M = r (+8 for hi half)
    x[r] = xn[(size_t)brow[r] * NN + n];
  }
  // packed-h store addresses for this lane's (b, n) elements
  const int kts = n >> 5, kls = n & 31;
  const int halfs = (kls >> 3) & 1;
  const int elems = (kls & 7) + ((kls >> 4) << 3);
  size_t hoff[8];
#pragma unroll
  for (int r = 0; r < 8; ++r) {
    int lanew = (brow[r] & 15) + (halfs << 4);
    hoff[r] = ((size_t)(mt * KT_N + kts) * 32 + lanew) * 16 + elems;
  }
  // u-drive pack mapping (each WG writes its own mt slice; duplicates identical)
  const int ubl = tid >> 4, ui = tid & 15;
  const int ub_lane = ubl + (((ui >> 3) & 1) << 4);
  const size_t uoff = ((size_t)(mt * 32 + ub_lane)) * 16 + (ui & 7);
  const int ubatch = mt * 16 + ubl;

  const bf16* wbase = wpk + (size_t)nt * KT_TOT * FRAG_ELEMS + (size_t)lane * 16;
  const bf16* ufrag = ub + (size_t)mt * FRAG_ELEMS + (size_t)lane * 16;
  const bf16* alane = As + (size_t)lane * 16;

  for (int s = 0; s < TT; ++s) {
    {   // pack u_n = inp + 0.01*eps_inp into the drive fragment (bf16)
      size_t gi = ((size_t)ubatch * TT + s) * INPD + ui;
      ub[uoff] = (bf16)(inp[gi] + INP_NOISE_F * eps_inp[gi]);
    }
    {   // cooperative stage: this WG's 64 KB A slice, global(L2) -> LDS
      const v16bf* src = (const v16bf*)(hb + (size_t)(s & 1) * HB_ELEMS +
                                        (size_t)mt * KT_N * FRAG_ELEMS);
      v16bf* dst = (v16bf*)As;
#pragma unroll
      for (int i = 0; i < 8; ++i)
        dst[tid + i * 256] = src[tid + i * 256];
    }
    __threadfence();
    __syncthreads();

    // D = [h, u, 1, 0] x [W_eff^T ; W_in^T ; b_rec ; 0]   (65 bf16 WMMAs)
    // Chunked by 4: cluster 8 global + 8 DS b128 loads, then 4 WMMAs, so one
    // L2 round-trip is amortized over 4 matrix ops (staggered waitcnts).
    v8f acc = {};
#pragma unroll 1
    for (int c = 0; c < KT_N / 4; ++c) {
      const bf16* bb = wbase + (size_t)c * 4 * FRAG_ELEMS;
      const bf16* ab = alane + (size_t)c * 4 * FRAG_ELEMS;
      __builtin_prefetch(bb + 8 * FRAG_ELEMS, 0, 3);   // 2 chunks ahead in W stream
      v16bf b0 = *(const v16bf*)(bb);
      v16bf b1 = *(const v16bf*)(bb + FRAG_ELEMS);
      v16bf b2 = *(const v16bf*)(bb + 2 * FRAG_ELEMS);
      v16bf b3 = *(const v16bf*)(bb + 3 * FRAG_ELEMS);
      v16bf a0 = *(const v16bf*)(ab);
      v16bf a1 = *(const v16bf*)(ab + FRAG_ELEMS);
      v16bf a2 = *(const v16bf*)(ab + 2 * FRAG_ELEMS);
      v16bf a3 = *(const v16bf*)(ab + 3 * FRAG_ELEMS);
      acc = wmma_bf16(a0, b0, acc);
      acc = wmma_bf16(a1, b1, acc);
      acc = wmma_bf16(a2, b2, acc);
      acc = wmma_bf16(a3, b3, acc);
    }
    {   // peeled drive tile
      v16bf a = *(const v16bf*)ufrag;
      v16bf b = *(const v16bf*)(wbase + (size_t)KT_N * FRAG_ELEMS);
      acc = wmma_bf16(a, b, acc);
    }

    // leaky update + noise, emit hn (fp32) and next-step packed h (bf16)
    bf16* hwr = hb + (size_t)((s + 1) & 1) * HB_ELEMS;
#pragma unroll
    for (int r = 0; r < 8; ++r) {
      size_t gi = ((size_t)brow[r] * TT + s) * NN + n;
      float xv = ONE_M_ALPHA * x[r] + ALPHA_F * acc[r] + ACT_NOISE_F * eps_act[gi];
      x[r] = xv;
      float h = fmaxf(xv, 0.f);
      hn[gi] = h;
      hwr[hoff[r]] = (bf16)h;
    }

    // device-wide barrier: monotonic counter, sense-free
    __threadfence();
    __syncthreads();
    if (tid == 0) {
      __hip_atomic_fetch_add(cnt, 1u, __ATOMIC_RELEASE, __HIP_MEMORY_SCOPE_AGENT);
      const unsigned target = (unsigned)NWG * (unsigned)(s + 1);
      while (__hip_atomic_load(cnt, __ATOMIC_ACQUIRE, __HIP_MEMORY_SCOPE_AGENT) < target)
        __builtin_amdgcn_s_sleep(2);
    }
    __syncthreads();
    __threadfence();
  }
}

// ---- WMMA readout: LDS-staged bf16 activations, 4 regions x 7 n-tiles -------
#define ROWW 1920   // columns 0..1919 cover pfc/acc/ofc/bla slices

__global__ void __launch_bounds__(256)
readout(const float* __restrict__ hn, const bf16* __restrict__ rpk,
        const float* __restrict__ bp, const float* __restrict__ ba,
        const float* __restrict__ bo, const float* __restrict__ bb_,
        float* __restrict__ out) {
  __shared__ bf16 hrow[16 * ROWW];                 // 61440 B
  const int tid = threadIdx.x, lane = tid & 31, wave = tid >> 5;
  const size_t row0 = (size_t)blockIdx.x * 16;

  for (int m = 0; m < 16; ++m)
    for (int c = tid; c < ROWW; c += 256)
      hrow[m * ROWW + c] = (bf16)hn[(row0 + m) * NN + c];
  __syncthreads();

  const int nl = lane & 15, hsel = lane >> 4;
  for (int p = wave; p < 28; p += 8) {             // (region, ntile) pairs
    const int rr = p / 7, ntile = p % 7;
    const int base = (rr == 3) ? 1408 : (rr << 9);
    const bf16* rbase = rpk + (size_t)((rr * 7 + ntile) * 16) * FRAG_ELEMS +
                        (size_t)lane * 16;
    const bf16* abase = &hrow[nl * ROWW + base + hsel * 8];
    v8f acc = {};
#pragma unroll 1
    for (int c = 0; c < 4; ++c) {                  // 4 chunks of 4 K-tiles
      union { v16bf v; v8bf h[2]; } a0, a1, a2, a3;
      const bf16* rb = rbase + (size_t)c * 4 * FRAG_ELEMS;
      const bf16* ab = abase + c * 4 * 32;
      v16bf b0 = *(const v16bf*)(rb);
      v16bf b1 = *(const v16bf*)(rb + FRAG_ELEMS);
      v16bf b2 = *(const v16bf*)(rb + 2 * FRAG_ELEMS);
      v16bf b3 = *(const v16bf*)(rb + 3 * FRAG_ELEMS);
      a0.h[0] = *(const v8bf*)(ab);        a0.h[1] = *(const v8bf*)(ab + 16);
      a1.h[0] = *(const v8bf*)(ab + 32);   a1.h[1] = *(const v8bf*)(ab + 48);
      a2.h[0] = *(const v8bf*)(ab + 64);   a2.h[1] = *(const v8bf*)(ab + 80);
      a3.h[0] = *(const v8bf*)(ab + 96);   a3.h[1] = *(const v8bf*)(ab + 112);
      acc = wmma_bf16(a0.v, b0, acc);
      acc = wmma_bf16(a1.v, b1, acc);
      acc = wmma_bf16(a2.v, b2, acc);
      acc = wmma_bf16(a3.v, b3, acc);
    }
    const int ncol = ntile * 16 + nl;
    if (ncol < UNITS) {
      const int ooff = (rr == 0) ? 200 : (rr == 1) ? 300 : (rr == 2) ? 0 : 100;
      const float* bias = (rr == 0) ? bp : (rr == 1) ? ba : (rr == 2) ? bo : bb_;
      const float bv = bias[ncol];
#pragma unroll
      for (int r = 0; r < 8; ++r) {
        int m = r + (hsel << 3);
        out[(row0 + m) * 400 + ooff + ncol] = acc[r] + bv;
      }
    }
  }
}

extern "C" void kernel_launch(void* const* d_in, const int* in_sizes, int n_in,
                              void* d_out, int out_size, void* d_ws, size_t ws_size,
                              hipStream_t stream) {
  const float* xn    = (const float*)d_in[0];
  const float* inp   = (const float*)d_in[1];
  const float* eps_i = (const float*)d_in[2];
  const float* eps_a = (const float*)d_in[3];
  const float* W_raw = (const float*)d_in[4];
  const float* W_in  = (const float*)d_in[5];
  const float* b_rec = (const float*)d_in[6];
  const float* W_pfc = (const float*)d_in[7];
  const float* b_pfc = (const float*)d_in[8];
  const float* W_acc = (const float*)d_in[9];
  const float* b_acc = (const float*)d_in[10];
  const float* W_ofc = (const float*)d_in[11];
  const float* b_ofc = (const float*)d_in[12];
  const float* W_bla = (const float*)d_in[13];
  const float* b_bla = (const float*)d_in[14];

  float* out = (float*)d_out;
  float* hn  = out + OUT_ELEMS;
  // Packed W_eff (8.5 MB bf16) lives in the out region until readout overwrites it.
  bf16* wpk = (bf16*)d_out;
  char* ws  = (char*)d_ws;
  unsigned* cnt = (unsigned*)(ws + OFF_CNT);
  bf16* hb  = (bf16*)(ws + OFF_HB);
  bf16* ub  = (bf16*)(ws + OFF_UB);
  bf16* rpk = (bf16*)(ws + OFF_RPK);

  (void)in_sizes; (void)n_in; (void)out_size; (void)ws_size;

  hipMemsetAsync(d_ws, 0, CLEAN_BYTES, stream);        // barrier counter + h/u buffers

  { int th = 128 * KT_TOT * 32;
    prep_w<<<(th + 255) / 256, 256, 0, stream>>>(W_raw, W_in, b_rec, wpk); }
  { int th = 4 * 7 * 16 * 32;
    prep_r<<<(th + 255) / 256, 256, 0, stream>>>(W_pfc, W_acc, W_ofc, W_bla, rpk); }
  { int th = BB * NN;
    init_h<<<(th + 255) / 256, 256, 0, stream>>>(xn, hb, ub); }

  rnn_recurrent<<<NWG, 256, 64 * 1024, stream>>>(xn, inp, eps_i, eps_a, wpk, hn, hb,
                                                 ub, cnt);
  readout<<<(BB * TT) / 16, 256, 0, stream>>>(hn, rpk, b_pfc, b_acc, b_ofc, b_bla, out);
}
